// Discriminator_61057255080565
// MI455X (gfx1250) — matrix operand order repaired and checked by reading, not compiled, not run on hardware
//
#include <hip/hip_runtime.h>

// ---------------------------------------------------------------------------
// Discriminator: LSTM(H=720) -> sigmoid -> FC(1024)+LeakyReLU -> FC(256)+LeakyReLU -> FC(1)
// CDNA5 (gfx1250): bf16 WMMA 16x16x32 (f32 accum), TDM async tile staging,
// M=128 workgroup tile with 2 A-fragments per wave for B-fragment reuse.
// ---------------------------------------------------------------------------

typedef __bf16 bf16;
typedef __attribute__((ext_vector_type(8)))  bf16  v8bf;
typedef __attribute__((ext_vector_type(16))) bf16  v16bf;
typedef __attribute__((ext_vector_type(8)))  float v8f;
typedef unsigned int uint32x4 __attribute__((ext_vector_type(4)));
typedef int          int32x4  __attribute__((ext_vector_type(4)));
typedef int          int32x8  __attribute__((ext_vector_type(8)));

constexpr int B_  = 2048;
constexpr int T_  = 256;
constexpr int F_  = 51;
constexpr int H_  = 720;
constexpr int HP  = 736;            // H padded to multiple of 32
constexpr int KP  = 800;            // 64 (padded x) + 736 (padded h)
constexpr int KT_STEP = KP / 32;    // 25 K-tiles in the recurrent GEMM
constexpr int JT   = HP / 32;       // 23 column tiles per gate
constexpr int NP   = 4 * HP;        // 2944 packed gate columns
constexpr int NTILES_WC = NP / 16;  // 184 B-tile slots (order: [nsub][gate])
constexpr int D1_ = 1024;
constexpr int D2_ = 256;
constexpr int KT1 = HP  / 32;       // 23
constexpr int KT2 = D1_ / 32;       // 32
constexpr float ALPHA_ = 0.3f;

constexpr int MROWS  = 128;                  // LSTM step workgroup M tile
constexpr int ABYTES = MROWS * 80;           // A tile: rows x (64B data + 16B pad)
constexpr int BBYTES = 8 * 1024;             // 8 packed B tiles
constexpr int BUFB   = ABYTES + BBYTES;      // 18432 (16-aligned)

constexpr int ABYTES_M = 64 * 80;            // MLP kernels keep the 64-row tile
constexpr int BUFB_M   = ABYTES_M + BBYTES;

#if defined(__gfx1250__) && __has_builtin(__builtin_amdgcn_tensor_load_to_lds)
#define HAS_TDM 1
#else
#define HAS_TDM 0
#endif

__device__ __forceinline__ float sigf(float x)   { return 1.0f / (1.0f + __expf(-x)); }
__device__ __forceinline__ float tanhft(float x) { return 1.0f - 2.0f / (__expf(2.0f * x) + 1.0f); }

__device__ __forceinline__ v16bf combine16(v8bf lo, v8bf hi) {
    v16bf r;
#pragma unroll
    for (int i = 0; i < 8; ++i) { r[i] = lo[i]; r[i + 8] = hi[i]; }
    return r;
}

__device__ __forceinline__ v8f wmma_bf16(v16bf a, v16bf b, v8f c) {
    return __builtin_amdgcn_wmma_f32_16x16x32_bf16(false, a, false, b, (short)0, c,
                                                   false, false);
}

// LSTM step compute: 2 A-fragments (32 rows) per wave, 8 B slots -> 16 wmma.
__device__ __forceinline__ void gemm_tiles2(const bf16* sA, const bf16* sB,
                                            v8f (&acc)[2][8], int w, int lane) {
    const bf16* aB0 = sA + (w * 32 + (lane & 15)) * 40 + ((lane >> 4) * 16);
    const bf16* aB1 = aB0 + 16 * 40;
    v16bf a0 = combine16(*(const v8bf*)aB0, *(const v8bf*)(aB0 + 8));
    v16bf a1 = combine16(*(const v8bf*)aB1, *(const v8bf*)(aB1 + 8));
#pragma unroll
    for (int nt = 0; nt < 8; ++nt) {
        const bf16* bB = sB + nt * 512 + lane * 16;
        v16bf bfrag = combine16(*(const v8bf*)bB, *(const v8bf*)(bB + 8));
        acc[0][nt] = wmma_bf16(a0, bfrag, acc[0][nt]);
        acc[1][nt] = wmma_bf16(a1, bfrag, acc[1][nt]);
    }
}

// MLP compute: 1 A-fragment (16 rows) per wave, 8 B slots -> 8 wmma.
__device__ __forceinline__ void gemm_tiles(const bf16* sA, const bf16* sB,
                                           v8f (&acc)[8], int w, int lane) {
    const bf16* aB = sA + (w * 16 + (lane & 15)) * 40 + ((lane >> 4) * 16);
    v16bf afrag = combine16(*(const v8bf*)aB, *(const v8bf*)(aB + 8));
#pragma unroll
    for (int nt = 0; nt < 8; ++nt) {
        const bf16* bB = sB + nt * 512 + lane * 16;
        v16bf bfrag = combine16(*(const v8bf*)bB, *(const v8bf*)(bB + 8));
        acc[nt] = wmma_bf16(afrag, bfrag, acc[nt]);
    }
}

#if HAS_TDM
// ---- Tensor Data Mover: 2D tile load, descriptor per CDNA5 ISA ----
__device__ __forceinline__ void tdm_issue(uint32x4 g0, int32x8 g1) {
    int32x4 z4 = {0, 0, 0, 0};
#if __clang_major__ >= 23
    int32x8 z8 = {0, 0, 0, 0, 0, 0, 0, 0};
    __builtin_amdgcn_tensor_load_to_lds(g0, g1, z4, z4, z8, 0);
#else
    __builtin_amdgcn_tensor_load_to_lds(g0, g1, z4, z4, 0);
#endif
}

// data_size = 3 (8-byte units); tile0/stride in 8B units; tensor dims == tile dims.
__device__ __forceinline__ void tdm_load_2d(const void* gptr, const void* lptr,
                                            unsigned tile0, unsigned tile1,
                                            unsigned long long stride0, unsigned flags) {
    unsigned long long ga = (unsigned long long)(uintptr_t)gptr;
    uint32x4 g0;
    g0[0] = 1u;                                                 // count=1, user mode
    g0[1] = (unsigned)(uintptr_t)lptr;                          // LDS offset (addr[31:0])
    g0[2] = (unsigned)(ga & 0xFFFFFFFFu);
    g0[3] = (unsigned)((ga >> 32) & 0x01FFFFFFu) | (2u << 30);  // addr[56:32] | type=2
    int32x8 g1;
    g1[0] = (int)flags;                                         // mask=0 | data_size | pad cfg
    g1[1] = (int)((tile0 & 0xFFFFu) << 16);                     // tensor_dim0[15:0]
    g1[2] = (int)(((tile0 >> 16) & 0xFFFFu) | ((tile1 & 0xFFFFu) << 16)); // tdim0 hi | tdim1 lo
    g1[3] = (int)(((tile1 >> 16) & 0xFFFFu) | ((tile0 & 0xFFFFu) << 16)); // tdim1 hi | tile_dim0
    g1[4] = (int)(tile1 & 0xFFFFu);                             // tile_dim1 | tile_dim2=0
    g1[5] = (int)(unsigned)(stride0 & 0xFFFFFFFFu);             // tensor_dim0_stride lo
    g1[6] = (int)(unsigned)((stride0 >> 32) & 0xFFFFu);         // stride hi | dim1_stride=0
    g1[7] = 0;
    tdm_issue(g0, g1);
}
constexpr unsigned TDM_FLAGS_B = 3u << 16;                          // 8B units
constexpr unsigned TDM_FLAGS_A = (3u << 16) | (1u << 20)            // pad_enable
                               | (3u << 22)                          // pad_interval: 16 DW (64B)
                               | (3u << 25);                         // pad_amount: 4 DW (16B)
#endif

// ---------------------------------------------------------------------------
// Weight packing into WMMA B-fragment tiles.
// Tile = 32(K) x 16(N), 1KB: lane l<16 -> (K=e, N=l); lane>=16 -> (K=16+e, N=l-16),
// stored [lane][16 bf16]. Slot order for Wc: slot = nsub*4 + gate so the 8 tiles a
// workgroup needs per K-iter are contiguous (one TDM 2D tile of 8 rows).
// ---------------------------------------------------------------------------

__global__ void pack_wc(const float* __restrict__ Wx, const float* __restrict__ Wh,
                        bf16* __restrict__ wc) {
    int idx = blockIdx.x * 256 + threadIdx.x;
    if (idx >= NTILES_WC * KT_STEP * 512) return;
    int within = idx & 511, tile = idx >> 9;
    int lane = within >> 4, e = within & 15;
    int kt = tile % KT_STEP, slot = tile / KT_STEP;
    int g = slot & 3, nsub = slot >> 2;
    int j = nsub * 16 + (lane & 15);
    int K = kt * 32 + ((lane < 16) ? e : 16 + e);
    float v = 0.0f;
    if (j < H_) {
        int col = g * H_ + j;                 // gate-major source column (i,f,c,o)
        if (K < 64) { if (K < F_) v = Wx[(size_t)K * (4 * H_) + col]; }
        else        { int kh = K - 64; if (kh < H_) v = Wh[(size_t)kh * (4 * H_) + col]; }
    }
    wc[idx] = (bf16)v;
}

__global__ void pack_b(const float* __restrict__ W, bf16* __restrict__ dst,
                       int ktiles, int Ksrc, int Nsrc, int total) {
    int idx = blockIdx.x * 256 + threadIdx.x;
    if (idx >= total) return;
    int within = idx & 511, tile = idx >> 9;
    int lane = within >> 4, e = within & 15;
    int kt = tile % ktiles, ntile = tile / ktiles;
    int N = ntile * 16 + (lane & 15);
    int K = kt * 32 + ((lane < 16) ? e : 16 + e);
    float v = (K < Ksrc && N < Nsrc) ? W[(size_t)K * Nsrc + N] : 0.0f;
    dst[idx] = (bf16)v;
}

// x -> bf16 padded [B][T][64] so A staging is branch-free vector loads.
__global__ void pack_x(const float* __restrict__ x, bf16* __restrict__ xbf) {
    int i = blockIdx.x * 256 + threadIdx.x;
    if (i >= B_ * T_ * 64) return;
    int k = i & 63, bt = i >> 6;
    xbf[i] = (bf16)((k < F_) ? x[(size_t)bt * F_ + k] : 0.0f);
}

__global__ void init_state(const float* __restrict__ b, float* __restrict__ biasp,
                           float* __restrict__ c, bf16* __restrict__ hbf) {
    int i = blockIdx.x * 256 + threadIdx.x;
    if (i < NP) {
        int g = i / HP, j = i % HP;
        biasp[i] = (j < H_) ? b[g * H_ + j] : 0.0f;
    }
    if (i < B_ * HP) { c[i] = 0.0f; hbf[i] = (bf16)0.0f; }
}

// ---------------------------------------------------------------------------
// One LSTM timestep: z = [x_t | h] @ Wc + b ; gate math ; update c, h.
// Grid (B/128, HP/32), block 128 = 4 wave32. Wave w owns rows [w*32, w*32+32).
// Slot nt -> gate = nt&3, sub = nt>>2.
// ---------------------------------------------------------------------------
__global__ __launch_bounds__(128) void lstm_step(
    const bf16* __restrict__ xbf, const bf16* __restrict__ wc,
    const float* __restrict__ biasp, float* __restrict__ c,
    bf16* __restrict__ hbf, int t) {

    const int tid = threadIdx.x;
    const int lane = tid & 31, w = tid >> 5;
    const int mbase = blockIdx.x * MROWS;
    const int jt = blockIdx.y;

    v8f acc[2][8];
#pragma unroll
    for (int m = 0; m < 2; ++m)
#pragma unroll
        for (int i = 0; i < 8; ++i)
#pragma unroll
            for (int e = 0; e < 8; ++e) acc[m][i][e] = 0.0f;

#if HAS_TDM
    __shared__ __align__(16) char smem[2 * BUFB];
    auto issue = [&](int kt, int buf) {
        const char* base = smem + buf * BUFB;
        // A tile: MROWS rows x 64B (x for kt<2, h otherwise), padded to 80B rows in LDS.
        const bf16* asrc;
        unsigned long long astride;
        if (kt * 32 < 64) {
            asrc = xbf + ((size_t)mbase * T_ + t) * 64 + kt * 32;
            astride = (unsigned long long)(T_ * 64 * 2 / 8);   // 4096 units
        } else {
            asrc = hbf + (size_t)mbase * HP + (kt * 32 - 64);
            astride = (unsigned long long)(HP * 2 / 8);        // 184 units
        }
        tdm_load_2d(asrc, base, 8, MROWS, astride, TDM_FLAGS_A);
        // B tile row: 8 contiguous slots x 1KB, slot stride = 25KB.
        const bf16* bsrc = wc + ((size_t)(jt * 8) * KT_STEP + kt) * 512;
        tdm_load_2d(bsrc, base + ABYTES, 128, 8,
                    (unsigned long long)(KT_STEP * 1024 / 8), TDM_FLAGS_B);
    };
    if (w == 0) { issue(0, 0); __builtin_amdgcn_s_wait_tensorcnt(0); }
    __syncthreads();
    int cur = 0;
    for (int kt = 0; kt < KT_STEP; ++kt) {
        if (w == 0 && kt + 1 < KT_STEP) issue(kt + 1, cur ^ 1);  // prefetch next via TDM
        gemm_tiles2((const bf16*)(smem + cur * BUFB),
                    (const bf16*)(smem + cur * BUFB + ABYTES), acc, w, lane);
        if (w == 0) __builtin_amdgcn_s_wait_tensorcnt(0);
        __syncthreads();
        cur ^= 1;
    }
#else
    __shared__ __align__(16) char smem[BUFB];
    bf16* sA = (bf16*)smem;
    bf16* sB = (bf16*)(smem + ABYTES);
    for (int kt = 0; kt < KT_STEP; ++kt) {
        __syncthreads();
        { // stage A: MROWS x 32 bf16; thread tid handles one full 64B row
            int kg0 = kt * 32;
            const bf16* src = (kg0 < 64)
                ? (xbf + ((size_t)(mbase + tid) * T_ + t) * 64 + kg0)
                : (hbf + (size_t)(mbase + tid) * HP + (kg0 - 64));
            bf16* dst = sA + tid * 40;
            const v8bf* p = (const v8bf*)src;
            v8bf* q = (v8bf*)dst;
            q[0] = p[0]; q[1] = p[1]; q[2] = p[2]; q[3] = p[3];
        }
        { // stage B: 8 contiguous slots of 1KB
            int nt = tid >> 4;
            size_t base = (size_t)((jt * 8 + nt) * KT_STEP + kt) * 512 + (size_t)(tid & 15) * 32;
            const v8bf* p = (const v8bf*)(wc + base);
            v8bf* q = (v8bf*)(sB + nt * 512 + (tid & 15) * 32);
            q[0] = p[0]; q[1] = p[1]; q[2] = p[2]; q[3] = p[3];
        }
        __syncthreads();
        gemm_tiles2(sA, sB, acc, w, lane);
    }
#endif

    // gate epilogue (fp32): c' = sig(f)*c + sig(i)*tanh(g); h = sig(o)*tanh(c')
#pragma unroll
    for (int m = 0; m < 2; ++m) {
        int r0 = mbase + w * 32 + m * 16 + ((lane & 16) ? 8 : 0);
#pragma unroll
        for (int s = 0; s < 2; ++s) {
            int col = jt * 32 + s * 16 + (lane & 15);
            float bi = biasp[col], bf_ = biasp[HP + col];
            float bg = biasp[2 * HP + col], bo = biasp[3 * HP + col];
#pragma unroll
            for (int v = 0; v < 8; ++v) {
                int r = r0 + v;
                float zi = acc[m][s * 4 + 0][v] + bi;
                float zf = acc[m][s * 4 + 1][v] + bf_;
                float zg = acc[m][s * 4 + 2][v] + bg;
                float zo = acc[m][s * 4 + 3][v] + bo;
                size_t idx = (size_t)r * HP + col;
                float cn = sigf(zf) * c[idx] + sigf(zi) * tanhft(zg);
                c[idx] = cn;
                hbf[idx] = (bf16)(sigf(zo) * tanhft(cn));
            }
        }
    }
}

// ---------------------------------------------------------------------------
// MLP layer 1: a1 = leaky(sigmoid(h) @ W1 + b1). Tile M=64 x N=128.
// ---------------------------------------------------------------------------
__global__ __launch_bounds__(128) void mlp1(const bf16* __restrict__ hbf,
                                            const bf16* __restrict__ w1p,
                                            const float* __restrict__ b1,
                                            bf16* __restrict__ a1) {
    __shared__ __align__(16) char smem[BUFB_M];
    bf16* sA = (bf16*)smem;
    bf16* sB = (bf16*)(smem + ABYTES_M);
    const int tid = threadIdx.x, lane = tid & 31, w = tid >> 5;
    const int mbase = blockIdx.x * 64;

    v8f acc[8];
#pragma unroll
    for (int i = 0; i < 8; ++i)
#pragma unroll
        for (int e = 0; e < 8; ++e) acc[i][e] = 0.0f;

    for (int kt = 0; kt < KT1; ++kt) {
        __syncthreads();
        { // stage A = sigmoid(h) in bf16
            int row = tid >> 1, half = tid & 1;
            int kg0 = kt * 32 + half * 16;
            const v8bf* p = (const v8bf*)(hbf + (size_t)(mbase + row) * HP + kg0);
            v8bf u0 = p[0], u1 = p[1];
            bf16* dst = sA + row * 40 + half * 16;
#pragma unroll
            for (int i = 0; i < 8; ++i) {
                dst[i]     = (bf16)sigf((float)u0[i]);
                dst[8 + i] = (bf16)sigf((float)u1[i]);
            }
        }
        {
            int nt = tid >> 4;
            int ntile = blockIdx.y * 8 + nt;
            size_t base = (size_t)(ntile * KT1 + kt) * 512 + (size_t)(tid & 15) * 32;
            const v8bf* p = (const v8bf*)(w1p + base);
            v8bf* q = (v8bf*)(sB + nt * 512 + (tid & 15) * 32);
            q[0] = p[0]; q[1] = p[1]; q[2] = p[2]; q[3] = p[3];
        }
        __syncthreads();
        gemm_tiles(sA, sB, acc, w, lane);
    }

    int r0 = mbase + w * 16 + ((lane & 16) ? 8 : 0);
    int colBase = blockIdx.y * 128 + (lane & 15);
#pragma unroll
    for (int nt = 0; nt < 8; ++nt) {
        int col = colBase + nt * 16;
        float bb = b1[col];
#pragma unroll
        for (int v = 0; v < 8; ++v) {
            float y = acc[nt][v] + bb;
            y = (y >= 0.0f) ? y : ALPHA_ * y;
            a1[(size_t)(r0 + v) * D1_ + col] = (bf16)y;
        }
    }
}

// ---------------------------------------------------------------------------
// MLP layer 2: a2 = leaky(a1 @ W2 + b2). Tile M=64 x N=128.
// ---------------------------------------------------------------------------
__global__ __launch_bounds__(128) void mlp2(const bf16* __restrict__ a1,
                                            const bf16* __restrict__ w2p,
                                            const float* __restrict__ b2,
                                            bf16* __restrict__ a2) {
    __shared__ __align__(16) char smem[BUFB_M];
    bf16* sA = (bf16*)smem;
    bf16* sB = (bf16*)(smem + ABYTES_M);
    const int tid = threadIdx.x, lane = tid & 31, w = tid >> 5;
    const int mbase = blockIdx.x * 64;

    v8f acc[8];
#pragma unroll
    for (int i = 0; i < 8; ++i)
#pragma unroll
        for (int e = 0; e < 8; ++e) acc[i][e] = 0.0f;

    for (int kt = 0; kt < KT2; ++kt) {
        __syncthreads();
        {
            int row = tid >> 1, half = tid & 1;
            int kg0 = kt * 32 + half * 16;
            const v8bf* p = (const v8bf*)(a1 + (size_t)(mbase + row) * D1_ + kg0);
            bf16* dst = sA + row * 40 + half * 16;
            ((v8bf*)dst)[0] = p[0];
            ((v8bf*)dst)[1] = p[1];
        }
        {
            int nt = tid >> 4;
            int ntile = blockIdx.y * 8 + nt;
            size_t base = (size_t)(ntile * KT2 + kt) * 512 + (size_t)(tid & 15) * 32;
            const v8bf* p = (const v8bf*)(w2p + base);
            v8bf* q = (v8bf*)(sB + nt * 512 + (tid & 15) * 32);
            q[0] = p[0]; q[1] = p[1]; q[2] = p[2]; q[3] = p[3];
        }
        __syncthreads();
        gemm_tiles(sA, sB, acc, w, lane);
    }

    int r0 = mbase + w * 16 + ((lane & 16) ? 8 : 0);
    int colBase = blockIdx.y * 128 + (lane & 15);
#pragma unroll
    for (int nt = 0; nt < 8; ++nt) {
        int col = colBase + nt * 16;
        float bb = b2[col];
#pragma unroll
        for (int v = 0; v < 8; ++v) {
            float y = acc[nt][v] + bb;
            y = (y >= 0.0f) ? y : ALPHA_ * y;
            a2[(size_t)(r0 + v) * D2_ + col] = (bf16)y;
        }
    }
}

// Final FC: out[b] = a2[b,:] . W3 + b3. One wave32 per row.
__global__ __launch_bounds__(128) void mlp3(const bf16* __restrict__ a2,
                                            const float* __restrict__ W3,
                                            const float* __restrict__ b3,
                                            float* __restrict__ out) {
    int row = blockIdx.x * 4 + (threadIdx.x >> 5);
    int lane = threadIdx.x & 31;
    float s = 0.0f;
#pragma unroll
    for (int k = lane; k < D2_; k += 32)
        s += (float)a2[(size_t)row * D2_ + k] * W3[k];
#pragma unroll
    for (int m = 16; m > 0; m >>= 1) s += __shfl_xor(s, m, 32);
    if (lane == 0) out[row] = s + b3[0];
}

// ---------------------------------------------------------------------------

extern "C" void kernel_launch(void* const* d_in, const int* in_sizes, int n_in,
                              void* d_out, int out_size, void* d_ws, size_t ws_size,
                              hipStream_t stream) {
    (void)in_sizes; (void)n_in; (void)out_size; (void)ws_size;
    const float* x  = (const float*)d_in[0];
    const float* Wx = (const float*)d_in[1];
    const float* Wh = (const float*)d_in[2];
    const float* b  = (const float*)d_in[3];
    const float* W1 = (const float*)d_in[4];
    const float* b1 = (const float*)d_in[5];
    const float* W2 = (const float*)d_in[6];
    const float* b2 = (const float*)d_in[7];
    const float* W3 = (const float*)d_in[8];
    const float* b3 = (const float*)d_in[9];
    float* out = (float*)d_out;

    char* ws = (char*)d_ws;
    size_t off = 0;
    auto take = [&](size_t bytes) -> char* {
        char* p = ws + off;
        off = (off + bytes + 255) & ~(size_t)255;
        return p;
    };
    bf16*  wc    = (bf16*) take((size_t)NTILES_WC * KT_STEP * 512 * 2);
    bf16*  w1p   = (bf16*) take((size_t)(D1_ / 16) * KT1 * 512 * 2);
    bf16*  w2p   = (bf16*) take((size_t)(D2_ / 16) * KT2 * 512 * 2);
    float* biasp = (float*)take((size_t)NP * 4);
    float* c     = (float*)take((size_t)B_ * HP * 4);
    bf16*  hbf   = (bf16*) take((size_t)B_ * HP * 2);
    bf16*  a1    = (bf16*) take((size_t)B_ * D1_ * 2);
    bf16*  a2    = (bf16*) take((size_t)B_ * D2_ * 2);
    bf16*  xbf   = (bf16*) take((size_t)B_ * T_ * 64 * 2);

    { int tot = NTILES_WC * KT_STEP * 512;
      pack_wc<<<(tot + 255) / 256, 256, 0, stream>>>(Wx, Wh, wc); }
    { int tot = (D1_ / 16) * KT1 * 512;
      pack_b<<<(tot + 255) / 256, 256, 0, stream>>>(W1, w1p, KT1, H_, D1_, tot); }
    { int tot = (D2_ / 16) * KT2 * 512;
      pack_b<<<(tot + 255) / 256, 256, 0, stream>>>(W2, w2p, KT2, D1_, D2_, tot); }
    { int tot = B_ * T_ * 64;
      pack_x<<<(tot + 255) / 256, 256, 0, stream>>>(x, xbf); }
    { int tot = B_ * HP;
      init_state<<<(tot + 255) / 256, 256, 0, stream>>>(b, biasp, c, hbf); }

    dim3 gs(B_ / MROWS, JT);
    for (int t = 0; t < T_; ++t)
        lstm_step<<<gs, 128, 0, stream>>>(xbf, wc, biasp, c, hbf, t);

    mlp1<<<dim3(B_ / 64, D1_ / 128), 128, 0, stream>>>(hbf, w1p, b1, a1);
    mlp2<<<dim3(B_ / 64, D2_ / 128), 128, 0, stream>>>(a1, w2p, b2, a2);
    mlp3<<<B_ / 4, 128, 0, stream>>>(a2, W3, b3, out);
}
